// Model_11330123727132
// MI455X (gfx1250) — compile-verified, matrix operand order checked by reading, not run on hardware
//
#include <hip/hip_runtime.h>

// ---------------------------------------------------------------------------
// Problem constants (match reference)
// ---------------------------------------------------------------------------
constexpr int N    = 2048;
constexpr int D    = 128;
constexpr int K    = 16;
constexpr int T    = 16;
constexpr int E    = 8192;
constexpr int TWOE = 2 * E;        // 16384
constexpr float GAMMA = 0.1f;
constexpr float SIGMA = 1.0f;
constexpr float EPSC  = 1e-10f;

// ---------------------------------------------------------------------------
// WMMA types / helpers (CDNA5 gfx1250, wave32)
// ---------------------------------------------------------------------------
typedef __attribute__((ext_vector_type(16))) __bf16 v16bf;
typedef __attribute__((ext_vector_type(8)))  float  v8f;

__device__ __forceinline__ v8f wmma_bf16(v16bf a, v16bf b, v8f c) {
  // D = A(16x32 bf16) * B(32x16 bf16) + C(16x16 f32)
  return __builtin_amdgcn_wmma_f32_16x16x32_bf16(
      /*neg_a=*/false, a, /*neg_b=*/false, b,
      /*c_mod=*/(short)0, c, /*reuse_a=*/false, /*reuse_b=*/false);
}

// Per-lane fragment pattern (A rows / B cols indexed by lane&15):
//   lane<16 : elements 0-7 -> K = kbase+0..7,   elements 8-15 -> K = kbase+16..23
//   lane>=16: elements 0-7 -> K = kbase+8..15,  elements 8-15 -> K = kbase+24..31
// i.e. klo = (lane>>4)*8; halves are 8 contiguous bf16 = one b128 load.
__device__ __forceinline__ v16bf frag_bf16(const __bf16* row, int kbase, int klo) {
  union { uint4 q[2]; v16bf v; } u;
  u.q[0] = *reinterpret_cast<const uint4*>(row + kbase + klo);
  u.q[1] = *reinterpret_cast<const uint4*>(row + kbase + 16 + klo);
  return u.v;
}
__device__ __forceinline__ v16bf frag_f32(const float* row, int kbase, int klo) {
  v16bf r;
#pragma unroll
  for (int i = 0; i < 8; ++i) r[i]     = (__bf16)row[kbase + klo + i];
#pragma unroll
  for (int i = 0; i < 8; ++i) r[i + 8] = (__bf16)row[kbase + 16 + klo + i];
  return r;
}
__device__ __forceinline__ v16bf frag_prod_f32(const float* ra, const float* rb,
                                               int kbase, int klo) {
  v16bf r;
#pragma unroll
  for (int i = 0; i < 8; ++i) {
    int k = kbase + klo + i;
    r[i] = (__bf16)(ra[k] * rb[k]);
  }
#pragma unroll
  for (int i = 0; i < 8; ++i) {
    int k = kbase + 16 + klo + i;
    r[i + 8] = (__bf16)(ra[k] * rb[k]);
  }
  return r;
}

__device__ __forceinline__ float sigmoidf_(float x) { return 1.0f / (1.0f + __expf(-x)); }
__device__ __forceinline__ float softplusf_(float x) {
  return (x > 20.0f) ? x : log1pf(__expf(x));
}

// ---------------------------------------------------------------------------
// gfx1250 async copy global->LDS (ASYNCcnt path); inline asm since no
// probe-confirmed builtin exists for these on either toolchain.
// ---------------------------------------------------------------------------
typedef __attribute__((address_space(3))) const unsigned short lds_cushort;

__device__ __forceinline__ unsigned lds_off_of(const unsigned short* p) {
  return (unsigned)(size_t)(lds_cushort*)p;   // generic -> LDS addrspace -> 32-bit offset
}
__device__ __forceinline__ void async_load_b128_to_lds(unsigned lds_off, const void* gaddr) {
  asm volatile("global_load_async_to_lds_b128 %0, %1, off"
               :: "v"(lds_off), "v"(gaddr)
               : "memory");
}
__device__ __forceinline__ void wait_asynccnt0() {
  asm volatile("s_wait_asynccnt 0" ::: "memory");
}

// ---------------------------------------------------------------------------
// Small utility kernels
// ---------------------------------------------------------------------------
__global__ void zero_kernel(float* p, int n) {
  int i = blockIdx.x * blockDim.x + threadIdx.x;
  if (i < n) p[i] = 0.0f;
}

__global__ void tobf16_kernel(const float* in, unsigned short* out, int n) {
  int i = blockIdx.x * blockDim.x + threadIdx.x;
  if (i < n) reinterpret_cast<__bf16*>(out)[i] = (__bf16)in[i];
}

// a_m, a_s and kld_alpha (one block of 128 threads)
__global__ __launch_bounds__(128)
void alpha_kernel(const float* mean_emb, const float* std_emb, const int* subj,
                  float* am, float* as_, float* accum) {
  int d = threadIdx.x;
  int si = subj[0];
  float m = mean_emb[si * D + d];
  float s = softplusf_(std_emb[si * D + d]);
  am[d] = m;
  as_[d] = s;
  __shared__ float red[128];
  red[d] = 0.5f * (-2.0f * logf(s) + s * s + m * m - 1.0f);
  __syncthreads();
  for (int st = 64; st > 0; st >>= 1) {
    if (d < st) red[d] += red[d + st];
    __syncthreads();
  }
  if (d == 0) accum[2] = red[0];  // kld_alpha (single timestep value)
}

// out[i] = b[i] + dot(W[i,:], am)   (W is [size, D])
__global__ __launch_bounds__(256)
void vec_linear_kernel(const float* W, const float* b, const float* am,
                       float* out, int size) {
  __shared__ float as_[D];
  int tid = threadIdx.x;
  if (tid < D) as_[tid] = am[tid];
  __syncthreads();
  int i = blockIdx.x * blockDim.x + tid;
  if (i >= size) return;
  const float* w = W + (size_t)i * D;
  float acc = b[i];
#pragma unroll 8
  for (int j = 0; j < D; ++j) acc += w[j] * as_[j];
  out[i] = acc;
}

// ---------------------------------------------------------------------------
// GRU cell: x = concat(prior, prior); gate order r, z, n (torch convention)
// one block per row (128 threads = feature dim)
// ---------------------------------------------------------------------------
__global__ __launch_bounds__(128)
void gru_kernel(const float* prior, const float* h_in, float* h_out,
                const float* Wih, const float* Whh,
                const float* bih, const float* bhh) {
  int n = blockIdx.x, d = threadIdx.x;
  __shared__ float xs[D], hs[D];
  xs[d] = prior[n * D + d];
  hs[d] = h_in[n * D + d];
  __syncthreads();

  float ir = bih[d], iz = bih[D + d], inn = bih[2 * D + d];
  const float* w0 = Wih + (size_t)d * (2 * D);
  const float* w1 = Wih + (size_t)(D + d) * (2 * D);
  const float* w2 = Wih + (size_t)(2 * D + d) * (2 * D);
#pragma unroll 4
  for (int j = 0; j < D; ++j) {
    float xv = xs[j];
    ir  += xv * (w0[j] + w0[D + j]);
    iz  += xv * (w1[j] + w1[D + j]);
    inn += xv * (w2[j] + w2[D + j]);
  }
  float hr = bhh[d], hz = bhh[D + d], hn = bhh[2 * D + d];
  const float* v0 = Whh + (size_t)d * D;
  const float* v1 = Whh + (size_t)(D + d) * D;
  const float* v2 = Whh + (size_t)(2 * D + d) * D;
#pragma unroll 4
  for (int j = 0; j < D; ++j) {
    float hv = hs[j];
    hr += hv * v0[j];
    hz += hv * v1[j];
    hn += hv * v2[j];
  }
  float r  = sigmoidf_(ir + hr);
  float zz = sigmoidf_(iz + hz);
  float nn = tanhf(inn + r * hn);
  h_out[n * D + d] = (1.0f - zz) * nn + zz * hs[d];
}

// ---------------------------------------------------------------------------
// mean/std/sample + KL(N(mean,std) || N(prior, s2)) accumulation
// one block per row
// ---------------------------------------------------------------------------
__global__ __launch_bounds__(128)
void linstat_kernel(const float* h, const float* prior,
                    const float* Wm, const float* bm,
                    const float* Ws, const float* bs,
                    const float* eps, float* samp_out,
                    float* accum, int acc_slot, float s2) {
  int n = blockIdx.x, d = threadIdx.x;
  __shared__ float hs[D];
  __shared__ float red[128];
  hs[d] = h[n * D + d];
  __syncthreads();
  const float* wm = Wm + (size_t)d * D;
  const float* ws = Ws + (size_t)d * D;
  float mean = bm[d], sv = bs[d];
#pragma unroll 4
  for (int j = 0; j < D; ++j) {
    float hv = hs[j];
    mean += hv * wm[j];
    sv   += hv * ws[j];
  }
  float stdv = softplusf_(sv);
  float pr   = prior[n * D + d];
  samp_out[n * D + d] = mean + stdv * eps[n * D + d];
  float dm = mean - pr;
  red[d] = 0.5f * (2.0f * (logf(s2) - logf(stdv)) +
                   (stdv * stdv + dm * dm) / (s2 * s2) - 1.0f);
  __syncthreads();
  for (int st = 64; st > 0; st >>= 1) {
    if (d < st) red[d] += red[d + st];
    __syncthreads();
  }
  if (d == 0) atomicAdd(&accum[acc_slot], red[0]);
}

// ---------------------------------------------------------------------------
// Edge kernel: each wave handles 16 edges (8 waves / block).
//   logits_q = (phi_s[w] * phi_s[c]) @ W_pi.T   [16 x K]   (WMMA)
//   prior    =  phi_s[w] @ beta_s.T             [16 x K]   (WMMA)
//   log-softmaxes, kld_z, gumbel-softmax z -> z_out
// ---------------------------------------------------------------------------
__global__ __launch_bounds__(256)
void edge_kernel(const int* edges, const float* u_gumbel,
                 const float* phi_s, const float* beta_s, const float* W_pi,
                 float* z_out, float* accum, int t) {
  const int lane  = threadIdx.x & 31;
  const int wv    = threadIdx.x >> 5;
  const int e0    = (blockIdx.x * 8 + wv) * 16;
  const int rl    = lane & 15;
  const int klo   = (lane >> 4) * 8;
  const int rbase = (lane >> 4) * 8;

  const int* et = edges + (size_t)t * E * 2;
  int j = e0 + rl;
  int wi, ci;
  if (j < E) { wi = et[j * 2 + 0]; ci = et[j * 2 + 1]; }
  else       { wi = et[(j - E) * 2 + 1]; ci = et[(j - E) * 2 + 0]; }

  const float* pw    = phi_s + (size_t)wi * D;
  const float* pc    = phi_s + (size_t)ci * D;
  const float* picol = W_pi   + (size_t)rl * D;   // B column rl = W_pi[rl,:]
  const float* bcol  = beta_s + (size_t)rl * D;   // B column rl = beta_s[rl,:]

  v8f cq = {0.f, 0.f, 0.f, 0.f, 0.f, 0.f, 0.f, 0.f};
  v8f cp = {0.f, 0.f, 0.f, 0.f, 0.f, 0.f, 0.f, 0.f};
#pragma unroll
  for (int kb = 0; kb < 4; ++kb) {
    v16bf au  = frag_prod_f32(pw, pc, 32 * kb, klo);
    v16bf apw = frag_f32(pw, 32 * kb, klo);
    v16bf bpi = frag_f32(picol, 32 * kb, klo);
    v16bf bbt = frag_f32(bcol, 32 * kb, klo);
    cq = wmma_bf16(au, bpi, cq);
    cp = wmma_bf16(apw, bbt, cp);
  }

  float kz = 0.0f;
#pragma unroll
  for (int v = 0; v < 8; ++v) {
    // C layout: VGPR v -> row = rbase+v ; lane&15 -> col
    float xq = cq[v];
    float mq = xq;
#pragma unroll
    for (int msk = 1; msk < 16; msk <<= 1) mq = fmaxf(mq, __shfl_xor(mq, msk, 32));
    float sq = __expf(xq - mq);
#pragma unroll
    for (int msk = 1; msk < 16; msk <<= 1) sq += __shfl_xor(sq, msk, 32);
    float logpost = xq - mq - logf(sq);
    float post = __expf(logpost);

    float xp = cp[v];
    float mp = xp;
#pragma unroll
    for (int msk = 1; msk < 16; msk <<= 1) mp = fmaxf(mp, __shfl_xor(mp, msk, 32));
    float sp = __expf(xp - mp);
#pragma unroll
    for (int msk = 1; msk < 16; msk <<= 1) sp += __shfl_xor(sp, msk, 32);
    float logprior = xp - mp - logf(sp);

    kz += post * (logpost - logprior);

    // gumbel-softmax z (TEMP == 1)
    int e = e0 + rbase + v;
    float u = u_gumbel[((size_t)t * TWOE + e) * K + rl];
    float g = -logf(-logf(u + EPSC) + EPSC);
    float xz = xq + g;
    float mz = xz;
#pragma unroll
    for (int msk = 1; msk < 16; msk <<= 1) mz = fmaxf(mz, __shfl_xor(mz, msk, 32));
    float ez = __expf(xz - mz);
    float sz = ez;
#pragma unroll
    for (int msk = 1; msk < 16; msk <<= 1) sz += __shfl_xor(sz, msk, 32);
    z_out[(size_t)e * K + rl] = ez / sz;
  }
#pragma unroll
  for (int msk = 1; msk < 32; msk <<= 1) kz += __shfl_xor(kz, msk, 32);
  if (lane == 0) atomicAdd(&accum[1], kz);
}

// ---------------------------------------------------------------------------
// y = z @ beta_s  ->  bf16  [TWOE, D]
// ---------------------------------------------------------------------------
__global__ __launch_bounds__(128)
void y_kernel(const float* z, const float* beta_s, unsigned short* ybf) {
  int e = blockIdx.x, d = threadIdx.x;
  __shared__ float zs[K];
  if (d < K) zs[d] = z[(size_t)e * K + d];
  __syncthreads();
  float acc = 0.0f;
#pragma unroll
  for (int k = 0; k < K; ++k) acc += zs[k] * beta_s[(size_t)k * D + d];
  reinterpret_cast<__bf16*>(ybf)[(size_t)e * D + d] = (__bf16)acc;
}

// ---------------------------------------------------------------------------
// Recon kernel (dominant GEMM): logits[e, n] = dot(y[e,:], W_dec[n,:])
// 8 waves/block, each wave owns 16 rows. W_dec tiles (16 cols x 128 K, bf16,
// 4KB) are staged in LDS once per BLOCK with gfx1250 async global->LDS copies
// (ASYNCcnt), double-buffered so tile t+1 DMA overlaps tile t WMMAs.
// Streaming logsumexp over the 2048 columns + target-logit capture -> nll.
// ---------------------------------------------------------------------------
__global__ __launch_bounds__(256)
void recon_kernel(const int* edges, const unsigned short* ybf_,
                  const unsigned short* wdecbf_, float* accum, int t) {
  __shared__ __align__(16) unsigned short btile[2][16 * D];  // 2 x 4KB

  const __bf16* ybf = reinterpret_cast<const __bf16*>(ybf_);
  const int tid   = threadIdx.x;
  const int lane  = tid & 31;
  const int wv    = tid >> 5;
  const int e0    = (blockIdx.x * 8 + wv) * 16;
  const int rl    = lane & 15;
  const int klo   = (lane >> 4) * 8;
  const int rbase = (lane >> 4) * 8;

  const unsigned lds0 = lds_off_of(&btile[0][0]) + (unsigned)tid * 16u;
  const unsigned lds1 = lds_off_of(&btile[1][0]) + (unsigned)tid * 16u;

  // kick off tile 0 DMA (each thread moves 16B; 256 threads = full 4KB tile)
  async_load_b128_to_lds(lds0, wdecbf_ + (size_t)tid * 8);

  // A fragments: rows of y (held in VGPRs for the entire column sweep)
  v16bf a[4];
  const __bf16* arow = ybf + (size_t)(e0 + rl) * D;
#pragma unroll
  for (int kb = 0; kb < 4; ++kb) a[kb] = frag_bf16(arow, 32 * kb, klo);

  // target columns (c index per owned row)
  const int* et = edges + (size_t)t * E * 2;
  int cc[8];
#pragma unroll
  for (int v = 0; v < 8; ++v) {
    int j = e0 + rbase + v;
    cc[v] = (j < E) ? et[j * 2 + 1] : et[(j - E) * 2 + 0];
  }

  float m[8], s[8], tg[8];
#pragma unroll
  for (int v = 0; v < 8; ++v) { m[v] = -1.0e30f; s[v] = 0.0f; tg[v] = 0.0f; }

  wait_asynccnt0();
  __syncthreads();

  int cur = 0;
  constexpr int NT = N / 16;  // 128 tiles
  for (int tile = 0; tile < NT; ++tile) {
    // prefetch next tile into the other LDS buffer (overlaps WMMAs below)
    if (tile + 1 < NT) {
      unsigned ldsn = cur ? lds0 : lds1;
      async_load_b128_to_lds(ldsn,
          wdecbf_ + (size_t)(tile + 1) * 16 * D + (size_t)tid * 8);
    }

    const __bf16* brow =
        reinterpret_cast<const __bf16*>(&btile[cur][rl * D]);
    v8f c = {0.f, 0.f, 0.f, 0.f, 0.f, 0.f, 0.f, 0.f};
#pragma unroll
    for (int kb = 0; kb < 4; ++kb)
      c = wmma_bf16(a[kb], frag_bf16(brow, 32 * kb, klo), c);

    int col = tile * 16 + rl;
#pragma unroll
    for (int v = 0; v < 8; ++v) {
      float x = c[v];
      if (x > m[v]) { s[v] = s[v] * __expf(m[v] - x) + 1.0f; m[v] = x; }
      else          { s[v] += __expf(x - m[v]); }
      if (col == cc[v]) tg[v] = x;
    }

    wait_asynccnt0();   // next tile landed
    __syncthreads();    // everyone done reading btile[cur]
    cur ^= 1;
  }

  // combine logsumexp across the 16 lanes that share each row
  float nll = 0.0f;
#pragma unroll
  for (int v = 0; v < 8; ++v) {
    float mv = m[v], sv = s[v], tv = tg[v];
#pragma unroll
    for (int msk = 1; msk < 16; msk <<= 1) {
      float om = __shfl_xor(mv, msk, 32);
      float os = __shfl_xor(sv, msk, 32);
      float nm = fmaxf(mv, om);
      sv = sv * __expf(mv - nm) + os * __expf(om - nm);
      mv = nm;
      tv += __shfl_xor(tv, msk, 32);
    }
    nll += mv + logf(sv) - tv;
  }
  nll += __shfl_xor(nll, 16, 32);  // rows 0-7 (half 0) + rows 8-15 (half 1)
  if (lane == 0) atomicAdd(&accum[0], nll);
}

// ---------------------------------------------------------------------------
// Finalize: out = [nll, kld_z, T*kld_alpha, kld_beta, kld_phi] / (T*2E)
// ---------------------------------------------------------------------------
__global__ void finalize_kernel(const float* accum, float* out) {
  if (threadIdx.x == 0 && blockIdx.x == 0) {
    const float cnt = (float)(T * TWOE);
    out[0] = accum[0] / cnt;
    out[1] = accum[1] / cnt;
    out[2] = (float)T * accum[2] / cnt;
    out[3] = accum[3] / cnt;
    out[4] = accum[4] / cnt;
  }
}

// ---------------------------------------------------------------------------
// Host orchestration
// ---------------------------------------------------------------------------
extern "C" void kernel_launch(void* const* d_in, const int* in_sizes, int n_in,
                              void* d_out, int out_size, void* d_ws, size_t ws_size,
                              hipStream_t stream) {
  (void)in_sizes; (void)n_in; (void)out_size; (void)ws_size;

  const int*   edges     = (const int*)d_in[0];
  const int*   subj      = (const int*)d_in[1];
  const float* eps_phi   = (const float*)d_in[2];
  const float* eps_beta  = (const float*)d_in[3];
  const float* u_gumbel  = (const float*)d_in[4];
  const float* a_mean    = (const float*)d_in[5];
  const float* a_std     = (const float*)d_in[6];
  const float* W_s2phi   = (const float*)d_in[7];
  const float* b_s2phi   = (const float*)d_in[8];
  const float* W_s2beta  = (const float*)d_in[9];
  const float* b_s2beta  = (const float*)d_in[10];
  const float* W_bmean   = (const float*)d_in[11];
  const float* b_bmean   = (const float*)d_in[12];
  const float* W_bstd    = (const float*)d_in[13];
  const float* b_bstd    = (const float*)d_in[14];
  const float* W_pmean   = (const float*)d_in[15];
  const float* b_pmean   = (const float*)d_in[16];
  const float* W_pstd    = (const float*)d_in[17];
  const float* b_pstd    = (const float*)d_in[18];
  const float* W_pi      = (const float*)d_in[19];
  const float* Wih_n     = (const float*)d_in[20];
  const float* Whh_n     = (const float*)d_in[21];
  const float* bih_n     = (const float*)d_in[22];
  const float* bhh_n     = (const float*)d_in[23];
  const float* Wih_c     = (const float*)d_in[24];
  const float* Whh_c     = (const float*)d_in[25];
  const float* bih_c     = (const float*)d_in[26];
  const float* bhh_c     = (const float*)d_in[27];
  const float* W_dec     = (const float*)d_in[28];
  float* out = (float*)d_out;

  // workspace carve-up (256B aligned)
  char* ws = (char*)d_ws;
  size_t off = 0;
  auto alloc = [&](size_t bytes) -> void* {
    void* p = ws + off;
    off += (bytes + 255) & ~(size_t)255;
    return p;
  };
  float* accum  = (float*)alloc(8 * sizeof(float));      // [nll, kldz, kldA, kldB, kldP]
  float* am     = (float*)alloc(D * sizeof(float));
  float* as_    = (float*)alloc(D * sizeof(float));
  float* phiA   = (float*)alloc((size_t)N * D * sizeof(float));
  float* phiB   = (float*)alloc((size_t)N * D * sizeof(float));
  float* hphiA  = (float*)alloc((size_t)N * D * sizeof(float));
  float* hphiB  = (float*)alloc((size_t)N * D * sizeof(float));
  float* betaA  = (float*)alloc((size_t)K * D * sizeof(float));
  float* betaB  = (float*)alloc((size_t)K * D * sizeof(float));
  float* hbetaA = (float*)alloc((size_t)K * D * sizeof(float));
  float* hbetaB = (float*)alloc((size_t)K * D * sizeof(float));
  float* zbuf   = (float*)alloc((size_t)TWOE * K * sizeof(float));
  unsigned short* ybf    = (unsigned short*)alloc((size_t)TWOE * D * 2);
  unsigned short* wdecbf = (unsigned short*)alloc((size_t)N * D * 2);

  // init: zeros for accumulators & GRU hidden states
  zero_kernel<<<1, 32, 0, stream>>>(accum, 8);
  zero_kernel<<<(N * D + 255) / 256, 256, 0, stream>>>(hphiA, N * D);
  zero_kernel<<<(K * D + 255) / 256, 256, 0, stream>>>(hbetaA, K * D);

  // alpha sample stats + kld_alpha
  alpha_kernel<<<1, 128, 0, stream>>>(a_mean, a_std, subj, am, as_, accum);

  // phi0 / beta0
  vec_linear_kernel<<<(N * D + 255) / 256, 256, 0, stream>>>(W_s2phi, b_s2phi, am, phiA, N * D);
  vec_linear_kernel<<<(K * D + 255) / 256, 256, 0, stream>>>(W_s2beta, b_s2beta, am, betaA, K * D);

  // W_dec -> bf16 (once per launch; deterministic)
  tobf16_kernel<<<(N * D + 255) / 256, 256, 0, stream>>>(W_dec, wdecbf, N * D);

  float* phi_prior  = phiA;  float* phi_s  = phiB;
  float* beta_prior = betaA; float* beta_s = betaB;
  float* hphi_in  = hphiA;  float* hphi_out  = hphiB;
  float* hbeta_in = hbetaA; float* hbeta_out = hbetaB;

  for (int t = 0; t < T; ++t) {
    // GRU updates
    gru_kernel<<<N, 128, 0, stream>>>(phi_prior, hphi_in, hphi_out,
                                      Wih_n, Whh_n, bih_n, bhh_n);
    gru_kernel<<<K, 128, 0, stream>>>(beta_prior, hbeta_in, hbeta_out,
                                      Wih_c, Whh_c, bih_c, bhh_c);

    // mean/std/sample + KL terms (slot 4 = kld_phi, slot 3 = kld_beta)
    linstat_kernel<<<N, 128, 0, stream>>>(hphi_out, phi_prior,
                                          W_pmean, b_pmean, W_pstd, b_pstd,
                                          eps_phi + (size_t)t * N * D, phi_s,
                                          accum, 4, SIGMA);
    linstat_kernel<<<K, 128, 0, stream>>>(hbeta_out, beta_prior,
                                          W_bmean, b_bmean, W_bstd, b_bstd,
                                          eps_beta + (size_t)t * K * D, beta_s,
                                          accum, 3, GAMMA);

    // per-edge topic logits, kld_z, gumbel-softmax z  (WMMA)
    edge_kernel<<<TWOE / (16 * 8), 256, 0, stream>>>(edges, u_gumbel, phi_s,
                                                     beta_s, W_pi, zbuf, accum, t);

    // y = z @ beta_s (bf16)
    y_kernel<<<TWOE, 128, 0, stream>>>(zbuf, beta_s, ybf);

    // dominant GEMM + logsumexp + NLL  (WMMA + async LDS staging)
    recon_kernel<<<TWOE / (16 * 8), 256, 0, stream>>>(edges, ybf, wdecbf, accum, t);

    // rotate carries
    { float* tmp = phi_prior;  phi_prior  = phi_s;  phi_s  = tmp; }
    { float* tmp = beta_prior; beta_prior = beta_s; beta_s = tmp; }
    { float* tmp = hphi_in;  hphi_in  = hphi_out;  hphi_out  = tmp; }
    { float* tmp = hbeta_in; hbeta_in = hbeta_out; hbeta_out = tmp; }
  }

  finalize_kernel<<<1, 32, 0, stream>>>(accum, out);
}
